// SparseBlock_25967372272142
// MI455X (gfx1250) — matrix operand order repaired and checked by reading, not compiled, not run on hardware
//
#include <hip/hip_runtime.h>

typedef __attribute__((ext_vector_type(16))) __bf16 v16bf;
typedef __attribute__((ext_vector_type(8)))  __bf16 v8bf;
typedef __attribute__((ext_vector_type(8)))  float  v8f;

#define N_VOX   262144
#define CIN     128
#define COUT    128
#define KOFF    27
#define LEAK    0.333f
#define EPSV    1e-4f

#define ROWS_PER_BLK 64
#define ROW_STRIDE   272   // 256B row + 16B pad -> conflict-free LDS b128

// ---------------- prep: W[k][ci][co] f32 -> Wt[k][co][ci] bf16 ----------------
// Also zero-fills the dummy "row N_VOX" of the bf16 feature array (gather target
// for invalid neighbors -> branchless staging in the conv kernel).
__global__ __launch_bounds__(256) void prep_w_kernel(const float* __restrict__ W,
                                                     __bf16* __restrict__ Wt,
                                                     __bf16* __restrict__ fb) {
  int i  = blockIdx.x * 256 + threadIdx.x;   // over 27*128*128
  int ci = i & 127;
  int co = (i >> 7) & 127;
  int k  = i >> 14;
  Wt[i] = (__bf16)W[((size_t)k * CIN + ci) * COUT + co];
  if (blockIdx.x == 0 && threadIdx.x < 16) {
    uint4 z = make_uint4(0u, 0u, 0u, 0u);
    *(uint4*)(fb + ((size_t)N_VOX << 7) + (threadIdx.x << 3)) = z;
  }
}

// ---------------- prep: feats f32 -> bf16 ----------------
__global__ __launch_bounds__(256) void prep_feats_kernel(const float* __restrict__ f,
                                                         __bf16* __restrict__ fb) {
  size_t i = ((size_t)blockIdx.x * 256 + threadIdx.x) * 8;
  float4 a = *(const float4*)(f + i);
  float4 b = *(const float4*)(f + i + 4);
  v8bf o;
  o[0] = (__bf16)a.x; o[1] = (__bf16)a.y; o[2] = (__bf16)a.z; o[3] = (__bf16)a.w;
  o[4] = (__bf16)b.x; o[5] = (__bf16)b.y; o[6] = (__bf16)b.z; o[7] = (__bf16)b.w;
  *(v8bf*)(fb + i) = o;
}

// ---------------- main sparse-conv GEMM via bf16 WMMA ----------------
// Block: 256 thr = 8 waves; 64 rows x 128 cols of output.
// wave w: row group rh = w>>1 (16 rows), co half cop = w&1 -> 4 co-tiles per wave.
// Software pipeline: prefetch k+1 gather into regs during k's WMMAs; gather is
// branchless (invalid neighbors read the zero row at index N_VOX).
__global__ __launch_bounds__(256) void conv_wmma_kernel(const __bf16* __restrict__ fb,
                                                        const __bf16* __restrict__ Wt,
                                                        const int*    __restrict__ nbr,
                                                        float*        __restrict__ out) {
  __shared__ __align__(16) unsigned char smem[ROWS_PER_BLK * ROW_STRIDE];
  const int t    = threadIdx.x;
  const int wave = t >> 5;
  const int lane = t & 31;
  const int lhi  = lane >> 4;     // 0: lanes 0-15, 1: lanes 16-31
  const int l15  = lane & 15;
  const int rh   = wave >> 1;     // 0..3
  const int cop  = wave & 1;      // 0..1
  const int row0 = blockIdx.x * ROWS_PER_BLK;

  // B fragment base: Wt[k][co][ci]; lane<16 takes ci 0..15 of chunk, lane>=16 takes 16..31
  const int co0 = cop * 64 + l15;
  const __bf16* wb = Wt + ((size_t)co0 << 7) + (lhi << 4);

  // A fragment: lane<16 -> row l15, K 0..15; lane>=16 -> row l15, K 16..31
  const unsigned abase = (unsigned)((rh * 16 + l15) * ROW_STRIDE + lhi * 32);

  v8f acc0 = {}, acc1 = {}, acc2 = {}, acc3 = {};

  uint4 vals[4];   // staging registers: 4 segments per thread

  // ---- prefetch k = 0 (branchless: invalid -> zero row N_VOX) ----
  #pragma unroll
  for (int j = 0; j < 4; ++j) {
    int sidx = t + 256 * j;
    int r = sidx >> 4, s = sidx & 15;
    int idx = nbr[(size_t)(row0 + r) * KOFF + 0];
    unsigned ridx = (idx >= 0) ? (unsigned)idx : (unsigned)N_VOX;
    vals[j] = *(const uint4*)(fb + ((size_t)ridx << 7) + (s << 3));
  }
  #pragma unroll
  for (int j = 0; j < 4; ++j) {
    int sidx = t + 256 * j;
    *(uint4*)(smem + (sidx >> 4) * ROW_STRIDE + (sidx & 15) * 16) = vals[j];
  }
  __syncthreads();

  #pragma unroll 1
  for (int k = 0; k < KOFF; ++k) {
    // ---- prefetch k+1 into registers (overlaps with WMMAs below) ----
    if (k + 1 < KOFF) {
      #pragma unroll
      for (int j = 0; j < 4; ++j) {
        int sidx = t + 256 * j;
        int r = sidx >> 4, s = sidx & 15;
        int idx = nbr[(size_t)(row0 + r) * KOFF + (k + 1)];
        unsigned ridx = (idx >= 0) ? (unsigned)idx : (unsigned)N_VOX;
        vals[j] = *(const uint4*)(fb + ((size_t)ridx << 7) + (s << 3));
      }
    }

    // ---- compute k: 4 ci-chunks x 4 co-tiles ----
    const __bf16* wk = wb + (size_t)k * (CIN * COUT);
    #pragma unroll
    for (int c = 0; c < 4; ++c) {
      v8bf lo = *(const v8bf*)(smem + abase + c * 64);
      v8bf hi = *(const v8bf*)(smem + abase + c * 64 + 16);
      v16bf a = __builtin_shufflevector(lo, hi, 0,1,2,3,4,5,6,7,8,9,10,11,12,13,14,15);
      v16bf b0 = *(const v16bf*)(wk + 0 * 2048 + c * 32);
      v16bf b1 = *(const v16bf*)(wk + 1 * 2048 + c * 32);
      v16bf b2 = *(const v16bf*)(wk + 2 * 2048 + c * 32);
      v16bf b3 = *(const v16bf*)(wk + 3 * 2048 + c * 32);
      acc0 = __builtin_amdgcn_wmma_f32_16x16x32_bf16(false, a, false, b0, (short)0, acc0, false, false);
      acc1 = __builtin_amdgcn_wmma_f32_16x16x32_bf16(false, a, false, b1, (short)0, acc1, false, false);
      acc2 = __builtin_amdgcn_wmma_f32_16x16x32_bf16(false, a, false, b2, (short)0, acc2, false, false);
      acc3 = __builtin_amdgcn_wmma_f32_16x16x32_bf16(false, a, false, b3, (short)0, acc3, false, false);
    }

    if (k + 1 < KOFF) {
      __syncthreads();   // all waves done reading buf for k
      #pragma unroll
      for (int j = 0; j < 4; ++j) {
        int sidx = t + 256 * j;
        *(uint4*)(smem + (sidx >> 4) * ROW_STRIDE + (sidx & 15) * 16) = vals[j];
      }
      __syncthreads();   // buf ready for k+1
    }
  }

  // D layout: VGPR r holds M=r (lanes 0-15) / M=r+8 (lanes 16-31), N = l15
  float* ob = out + ((size_t)(row0 + rh * 16 + lhi * 8)) * COUT + co0;
  #pragma unroll
  for (int r = 0; r < 8; ++r) {
    ob[(size_t)r * COUT]      = acc0[r];
    ob[(size_t)r * COUT + 16] = acc1[r];
    ob[(size_t)r * COUT + 32] = acc2[r];
    ob[(size_t)r * COUT + 48] = acc3[r];
  }
}

// ---------------- deterministic per-channel stats, stage 1 ----------------
__global__ __launch_bounds__(256) void stats1_kernel(const float* __restrict__ out,
                                                     float* __restrict__ psum,
                                                     float* __restrict__ psq) {
  __shared__ float ss[256], sq[256];
  const int b = blockIdx.x, t = threadIdx.x;
  const int c = t & 127, h = t >> 7;
  const float* p = out + ((size_t)b * 128 + h * 64) * COUT + c;
  float s = 0.f, q = 0.f;
  for (int r = 0; r < 64; ++r) { float v = p[(size_t)r * COUT]; s += v; q += v * v; }
  ss[t] = s; sq[t] = q;
  __syncthreads();
  if (t < 128) {
    psum[(size_t)b * 128 + t] = ss[t] + ss[t + 128];
    psq [(size_t)b * 128 + t] = sq[t] + sq[t + 128];
  }
}

// ---------------- stats stage 2: fold partials, compute scale/shift ----------------
__global__ void stats2_kernel(const float* __restrict__ psum, const float* __restrict__ psq,
                              const float* __restrict__ gamma, const float* __restrict__ beta,
                              float* __restrict__ scsh) {
  int c = threadIdx.x;    // 128 threads
  float s = 0.f, q = 0.f;
  for (int i = 0; i < 2048; ++i) {
    s += psum[(size_t)i * 128 + c];
    q += psq [(size_t)i * 128 + c];
  }
  float inv  = 1.0f / (float)N_VOX;
  float mean = s * inv;
  float var  = q * inv - mean * mean;
  float sc   = gamma[c] * rsqrtf(var + EPSV);
  scsh[c]       = sc;
  scsh[128 + c] = beta[c] - mean * sc;
}

// ---------------- BN + leaky ReLU, in place, float4 ----------------
__global__ __launch_bounds__(256) void bn_kernel(float* __restrict__ out,
                                                 const float* __restrict__ scsh) {
  size_t i = ((size_t)blockIdx.x * 256 + threadIdx.x) * 4;
  int c = (int)(i & 127);
  float4 v  = *(float4*)(out + i);
  float4 sc = *(const float4*)(scsh + c);
  float4 sh = *(const float4*)(scsh + 128 + c);
  float y;
  y = v.x * sc.x + sh.x; v.x = y > 0.f ? y : LEAK * y;
  y = v.y * sc.y + sh.y; v.y = y > 0.f ? y : LEAK * y;
  y = v.z * sc.z + sh.z; v.z = y > 0.f ? y : LEAK * y;
  y = v.w * sc.w + sh.w; v.w = y > 0.f ? y : LEAK * y;
  *(float4*)(out + i) = v;
}

extern "C" void kernel_launch(void* const* d_in, const int* in_sizes, int n_in,
                              void* d_out, int out_size, void* d_ws, size_t ws_size,
                              hipStream_t stream) {
  const float* feats = (const float*)d_in[0];
  const float* W     = (const float*)d_in[1];
  const float* gamma = (const float*)d_in[2];
  const float* beta  = (const float*)d_in[3];
  const int*   nbr   = (const int*)d_in[4];
  float* out = (float*)d_out;

  char* ws = (char*)d_ws;
  __bf16* Wt   = (__bf16*)ws;                              // 27*128*128*2 = 884736 B
  __bf16* fb   = (__bf16*)(ws + 884736);                   // (262144+1)*128*2 = 64 MiB + 256 B
  float*  psum = (float*)(ws + 884736 + 67109120);         // 2048*128*4 = 1 MiB
  float*  psq  = psum + 2048 * 128;                        // 1 MiB
  float*  scsh = psq  + 2048 * 128;                        // 256 floats

  prep_w_kernel    <<<(KOFF * CIN * COUT) / 256, 256, 0, stream>>>(W, Wt, fb);
  prep_feats_kernel<<<(N_VOX * CIN / 8) / 256,   256, 0, stream>>>(feats, fb);
  conv_wmma_kernel <<<N_VOX / ROWS_PER_BLK,      256, 0, stream>>>(fb, Wt, nbr, out);
  stats1_kernel    <<<2048,                      256, 0, stream>>>(out, psum, psq);
  stats2_kernel    <<<1,                         128, 0, stream>>>(psum, psq, gamma, beta, scsh);
  bn_kernel        <<<(N_VOX * COUT / 4) / 256,  256, 0, stream>>>(out, scsh);
}
